// CantorAttention_88983132439086
// MI455X (gfx1250) — compile-verified, matrix operand order checked by reading, not run on hardware
//
#include <hip/hip_runtime.h>
#include <hip/hip_bf16.h>
#include <stdint.h>

// ---------------------------------------------------------------------------
// CantorAttention for MI455X (gfx1250, wave32, WMMA).
//   1) x  (f32) -> bf16                           [cvt kernel]
//   2) w_qkv, w_out (f32, KxN) -> bf16 N-major    [transpose-convert]
//   3) qkv = x @ w_qkv + b_qkv                    [bf16 WMMA GEMM, f32 out]
//   4) routed attention, softmax, AV              [VALU, bf16 out]
//   5) out = attn @ w_out + b_out                 [bf16 WMMA GEMM, f32 out]
// GEMMs: double-buffered LDS tiles staged with GLOBAL_LOAD_ASYNC_TO_LDS_B128
// (ASYNCcnt) when the toolchain exposes it; sync b128 copies otherwise.
// ---------------------------------------------------------------------------

typedef __bf16    v16bf __attribute__((ext_vector_type(16)));
typedef float     v8f   __attribute__((ext_vector_type(8)));
typedef uint32_t  v4u   __attribute__((ext_vector_type(4)));
typedef uint32_t  v2u   __attribute__((ext_vector_type(2)));
typedef int       v4i   __attribute__((ext_vector_type(4)));

// AS-qualified pointer types for the async-LDS builtin
// (param 1: AS(1) "__device__" v4i*, param 2: AS(3) "__shared__" v4i*)
typedef __attribute__((address_space(1))) v4i* gas_v4i_ptr;
typedef __attribute__((address_space(3))) v4i* las_v4i_ptr;

static constexpr int Bc  = 2;
static constexpr int Sc  = 2048;
static constexpr int Dc  = 1024;
static constexpr int Hc  = 16;
static constexpr int Kc  = 64;    // routed keys per query
static constexpr int HDc = 64;    // head dim
static constexpr float SCALE = 0.125f; // 1/sqrt(64)

__device__ __forceinline__ uint32_t f32_to_bf16(float f) {
    union { float f; uint32_t u; } v; v.f = f;
    uint32_t u = v.u;
    uint32_t rnd = 0x7FFFu + ((u >> 16) & 1u);   // round-to-nearest-even
    return (u + rnd) >> 16;
}

// ------------------- async memory->LDS copy (16 B per lane) -----------------
#if __has_builtin(__builtin_amdgcn_global_load_async_to_lds_b128)
#define CANTOR_ASYNC_LDS 1
#else
#define CANTOR_ASYNC_LDS 0
#endif

__device__ __forceinline__ void copy16_to_lds(const void* g, void* l) {
#if CANTOR_ASYNC_LDS
    __builtin_amdgcn_global_load_async_to_lds_b128(
        (gas_v4i_ptr)g, (las_v4i_ptr)l, /*imm offset*/0, /*cpol*/0);
#else
    *(v4u*)l = *(const v4u*)g;
#endif
}

__device__ __forceinline__ void wait_async_copies() {
#if CANTOR_ASYNC_LDS
#if __has_builtin(__builtin_amdgcn_s_wait_asynccnt)
    __builtin_amdgcn_s_wait_asynccnt(0);
#else
    asm volatile("s_wait_asynccnt 0x0" ::: "memory");
#endif
#endif
}

// --------------------------- elementwise f32 -> bf16 ------------------------
__global__ __launch_bounds__(256)
void cantor_cvt_bf16(const float* __restrict__ in, unsigned short* __restrict__ out, int n4) {
    int i = blockIdx.x * blockDim.x + threadIdx.x;
    if (i < n4) {
        float4 v = ((const float4*)in)[i];
        uint32_t lo = f32_to_bf16(v.x) | (f32_to_bf16(v.y) << 16);
        uint32_t hi = f32_to_bf16(v.z) | (f32_to_bf16(v.w) << 16);
        v2u pk = { lo, hi };
        *(v2u*)(out + 4 * (size_t)i) = pk;
    }
}

// ---------------- transpose + convert: w[rows][cols] -> wt[cols][rows] ------
__global__ __launch_bounds__(256)
void cantor_transpose_bf16(const float* __restrict__ w, unsigned short* __restrict__ wt,
                           int rows, int cols) {
    __shared__ float tile[32][33];
    int bx = blockIdx.x * 32;   // col base
    int by = blockIdx.y * 32;   // row base
    int tx = threadIdx.x;       // 0..31
    int ty = threadIdx.y;       // 0..7
    #pragma unroll
    for (int j = 0; j < 32; j += 8)
        tile[ty + j][tx] = w[(size_t)(by + ty + j) * cols + bx + tx];
    __syncthreads();
    #pragma unroll
    for (int j = 0; j < 32; j += 8)
        wt[(size_t)(bx + ty + j) * rows + by + tx] =
            (unsigned short)f32_to_bf16(tile[tx][ty + j]);
}

// --------------------------- bf16 WMMA GEMM ---------------------------------
// C[M][N] = A[M][K] * Bt[N][K]^T + bias[N]   (A, Bt bf16; C f32)
// Block: 256 threads (8 wave32), tile 128x128, K-step 32, double-buffered LDS.
// Wave grid 4(m) x 2(n); each wave: 2x4 = 8 x v_wmma_f32_16x16x32_bf16.
static constexpr int BM = 128, BN = 128, BK = 32;
static constexpr int LDT = BK + 8;   // padded LDS row: 40 elems = 80 B (16B-mult)

union AFrag { v4u u[2]; v16bf v; };

__global__ __launch_bounds__(256)
void cantor_gemm_bf16(const unsigned short* __restrict__ A,
                      const unsigned short* __restrict__ Bt,
                      const float* __restrict__ bias,
                      float* __restrict__ C,
                      int M, int N, int Kd) {
    __shared__ unsigned short sA[2][BM * LDT];   // 2 x 10 240 B
    __shared__ unsigned short sB[2][BN * LDT];   // 2 x 10 240 B

    const int tid  = threadIdx.x;
    const int wave = tid >> 5;
    const int lane = tid & 31;
    const int m0 = blockIdx.y * BM;
    const int n0 = blockIdx.x * BN;
    const int wm = (wave & 3) * 32;   // wave row offset inside tile
    const int wn = (wave >> 2) * 64;  // wave col offset inside tile

    v8f acc[2][4] = {};

    const int lrow = tid >> 2;         // 0..63 (two passes cover 128 rows)
    const int lcol = (tid & 3) * 8;    // 0,8,16,24

    const int half = lane >> 4;        // 0/1
    const int lm   = lane & 15;        // fragment row (M or N)
    const int koff = half * 8;         // K chunk base per ISA 16-bit layout

    // stage one 128x32 A tile + 128x32 B tile into LDS buffer `buf`
    auto stage = [&](int buf, int k0) {
        #pragma unroll
        for (int p = 0; p < 2; ++p) {
            int r = lrow + p * 64;
            copy16_to_lds(A  + (size_t)(m0 + r) * Kd + k0 + lcol,
                          &sA[buf][r * LDT + lcol]);
            copy16_to_lds(Bt + (size_t)(n0 + r) * Kd + k0 + lcol,
                          &sB[buf][r * LDT + lcol]);
        }
    };

    stage(0, 0);
    wait_async_copies();
    __syncthreads();

    int buf = 0;
    for (int k0 = 0; k0 < Kd; k0 += BK) {
        // issue the next slab's copies into the other buffer (overlaps WMMA)
        if (k0 + BK < Kd) stage(buf ^ 1, k0 + BK);
        if (k0 + 2 * BK < Kd) {   // speculative prefetch two slabs ahead
            __builtin_prefetch(A  + (size_t)(m0 + lrow) * Kd + k0 + 2 * BK, 0, 0);
            __builtin_prefetch(Bt + (size_t)(n0 + lrow) * Kd + k0 + 2 * BK, 0, 0);
        }

        // ---- load fragments per ISA 16-bit A-matrix 16x32 layout ----
        AFrag fa[2], fb[4];
        #pragma unroll
        for (int i = 0; i < 2; ++i) {
            const unsigned short* p = &sA[buf][(wm + i * 16 + lm) * LDT + koff];
            fa[i].u[0] = *(const v4u*)p;            // K = koff .. koff+7
            fa[i].u[1] = *(const v4u*)(p + 16);     // K = koff+16 .. koff+23
        }
        #pragma unroll
        for (int j = 0; j < 4; ++j) {
            const unsigned short* p = &sB[buf][(wn + j * 16 + lm) * LDT + koff];
            fb[j].u[0] = *(const v4u*)p;
            fb[j].u[1] = *(const v4u*)(p + 16);
        }

        // ---- 8 WMMAs: D = A x B + C, f32 accumulate ----
        #pragma unroll
        for (int i = 0; i < 2; ++i)
            #pragma unroll
            for (int j = 0; j < 4; ++j)
                acc[i][j] = __builtin_amdgcn_wmma_f32_16x16x32_bf16(
                    false, fa[i].v, false, fb[j].v,
                    (short)0, acc[i][j], false, false);

        // next buffer must be fully landed before anyone reads it
        wait_async_copies();
        __syncthreads();
        buf ^= 1;
    }

    // ---- epilogue: C/D layout VGPR r -> M = r + 8*half, lane -> N = lm ----
    #pragma unroll
    for (int i = 0; i < 2; ++i) {
        #pragma unroll
        for (int j = 0; j < 4; ++j) {
            int col = n0 + wn + j * 16 + lm;
            float bv = bias[col];
            #pragma unroll
            for (int r = 0; r < 8; ++r) {
                int row = m0 + wm + i * 16 + half * 8 + r;
                C[(size_t)row * N + col] = acc[i][j][r] + bv;
            }
        }
    }
}

// --------------------------- routed attention -------------------------------
// One thread per (b,h,s): 64 gathered keys of dim 64, softmax, AV.
// q lives in registers; the 64 scores live in padded LDS so dynamic-k
// indexing never spills to scratch. Output written directly as bf16.
__global__ __launch_bounds__(128)
void cantor_attention(const float* __restrict__ qkv,      // (B*S, 3*D) f32
                      const int*   __restrict__ routes,   // (S, K)
                      unsigned short* __restrict__ out) { // (B*S, D) bf16
    __shared__ float sc[128][Kc + 1];   // +1 pad: lane stride 260 B, no bank clash

    int t = blockIdx.x * 128 + threadIdx.x;
    int s = t % Sc;
    int h = (t / Sc) % Hc;
    int b = t / (Sc * Hc);

    const size_t rowQ = (size_t)(b * Sc + s) * (3 * Dc);
    const float4* qv = (const float4*)(qkv + rowQ + h * HDc);

    float4 q[16];
    #pragma unroll
    for (int i = 0; i < 16; ++i) q[i] = qv[i];

    const int* rts = routes + (size_t)s * Kc;
    float* mysc = sc[threadIdx.x];

    // pass 1: scores + running max
    float mx = -3.0e38f;
    for (int k = 0; k < Kc; ++k) {
        int r = rts[k];
        const float4* kv = (const float4*)(qkv + (size_t)(b * Sc + r) * (3 * Dc) + Dc + h * HDc);
        float acc = 0.f;
        #pragma unroll
        for (int i = 0; i < 16; ++i) {
            float4 kk = kv[i];
            acc += q[i].x * kk.x + q[i].y * kk.y + q[i].z * kk.z + q[i].w * kk.w;
        }
        float sv = acc * SCALE;
        mysc[k] = sv;
        mx = fmaxf(mx, sv);
    }

    // pass 2: softmax normalizer
    float sum = 0.f;
    for (int k = 0; k < Kc; ++k) {
        float e = __expf(mysc[k] - mx);
        mysc[k] = e;
        sum += e;
    }
    const float inv = 1.0f / sum;

    // pass 3: AV accumulation
    float o[HDc];
    #pragma unroll
    for (int i = 0; i < HDc; ++i) o[i] = 0.f;
    for (int k = 0; k < Kc; ++k) {
        int r = rts[k];
        const float4* vv = (const float4*)(qkv + (size_t)(b * Sc + r) * (3 * Dc) + 2 * Dc + h * HDc);
        float p = mysc[k] * inv;
        #pragma unroll
        for (int i = 0; i < 16; ++i) {
            float4 vx = vv[i];
            o[4 * i + 0] += p * vx.x;
            o[4 * i + 1] += p * vx.y;
            o[4 * i + 2] += p * vx.z;
            o[4 * i + 3] += p * vx.w;
        }
    }

    unsigned short* orow = out + (size_t)(b * Sc + s) * Dc + h * HDc;
    #pragma unroll
    for (int i = 0; i < 16; ++i) {
        uint32_t lo = f32_to_bf16(o[4 * i + 0]) | (f32_to_bf16(o[4 * i + 1]) << 16);
        uint32_t hi = f32_to_bf16(o[4 * i + 2]) | (f32_to_bf16(o[4 * i + 3]) << 16);
        v2u pk = { lo, hi };
        *(v2u*)(orow + 4 * i) = pk;
    }
}

// --------------------------------- launcher ---------------------------------
extern "C" void kernel_launch(void* const* d_in, const int* in_sizes, int n_in,
                              void* d_out, int out_size, void* d_ws, size_t ws_size,
                              hipStream_t stream) {
    (void)in_sizes; (void)n_in; (void)out_size; (void)ws_size;

    const float* x      = (const float*)d_in[0];   // (B,S,D)
    const int*   routes = (const int*)  d_in[1];   // (S,K)
    const float* w_qkv  = (const float*)d_in[2];   // (D, 3D)
    const float* b_qkv  = (const float*)d_in[3];   // (3D)
    const float* w_out  = (const float*)d_in[4];   // (D, D)
    const float* b_out  = (const float*)d_in[5];   // (D)
    float*       out    = (float*)d_out;           // (B,S,D)

    char* ws = (char*)d_ws;
    const size_t MB = 1024 * 1024;
    unsigned short* x_bf    = (unsigned short*)(ws);              //  8 MB: (4096,1024) bf16
    unsigned short* wqkv_t  = (unsigned short*)(ws + 8  * MB);    //  6 MB: (3072,1024) bf16 N-major
    unsigned short* wout_t  = (unsigned short*)(ws + 14 * MB);    //  2 MB: (1024,1024) bf16 N-major
    float*          qkv     = (float*)        (ws + 16 * MB);     // 48 MB: (4096,3072) f32
    unsigned short* attn_o  = (unsigned short*)(ws + 64 * MB);    //  8 MB: (4096,1024) bf16

    const int MROWS = Bc * Sc;   // 4096

    // 1) x -> bf16
    {
        int n4 = (MROWS * Dc) / 4;   // 1 048 576
        cantor_cvt_bf16<<<n4 / 256, 256, 0, stream>>>(x, x_bf, n4);
    }
    // 2) weights -> bf16, N-major
    cantor_transpose_bf16<<<dim3((3 * Dc) / 32, Dc / 32), dim3(32, 8), 0, stream>>>(
        w_qkv, wqkv_t, Dc, 3 * Dc);
    cantor_transpose_bf16<<<dim3(Dc / 32, Dc / 32), dim3(32, 8), 0, stream>>>(
        w_out, wout_t, Dc, Dc);

    // 3) qkv = x @ w_qkv + b_qkv          (4096 x 3072 x 1024)
    cantor_gemm_bf16<<<dim3((3 * Dc) / BN, MROWS / BM), 256, 0, stream>>>(
        x_bf, wqkv_t, b_qkv, qkv, MROWS, 3 * Dc, Dc);

    // 4) routed attention                  (65536 queries)
    cantor_attention<<<(Bc * Hc * Sc) / 128, 128, 0, stream>>>(qkv, routes, attn_o);

    // 5) out = attn @ w_out + b_out        (4096 x 1024 x 1024)
    cantor_gemm_bf16<<<dim3(Dc / BN, MROWS / BM), 256, 0, stream>>>(
        attn_o, wout_t, b_out, out, MROWS, Dc, Dc);
}
